// top30_dropout_quant_decoder_36395552866330
// MI455X (gfx1250) — compile-verified, matrix operand order checked by reading, not run on hardware
//
#include <hip/hip_runtime.h>
#include <hip/hip_bf16.h>

typedef __attribute__((ext_vector_type(16))) _Float16 v16h;
typedef __attribute__((ext_vector_type(8)))  _Float16 v8h;
typedef __attribute__((ext_vector_type(8)))  float    v8f;

#define B_   128
#define C0   500
#define C0P  512
#define ZW   18        // 16 + halo(1) each side
#define O1   300
#define O1P  304
#define C1P  320       // h1 channel stride (300 padded)
#define H1P  40        // 34 + halo(2)+pad, offset +2
#define O2   100
#define O2P  112
#define C2P  128       // h2 channel stride (100 padded)
#define H2   36
#define O3P  16
#define K1   2048      // 4 taps * 512
#define TOPK 30

// ---- workspace layout (bytes) ----
#define OFF_L1  ((size_t)0)
#define OFF_Z   ((size_t)4096)
#define SZ_Z    ((size_t)B_*ZW*ZW*C0P*2)
#define OFF_H1  (OFF_Z + SZ_Z)
#define SZ_H1   ((size_t)B_*H1P*H1P*C1P*2)
#define OFF_H2  (OFF_H1 + SZ_H1)
#define SZ_H2   ((size_t)B_*H2*H2*C2P*2)
#define OFF_W1  (OFF_H2 + SZ_H2)
#define SZ_W1   ((size_t)4*O1P*K1*2)
#define OFF_W2  (OFF_W1 + SZ_W1)
#define SZ_W2   ((size_t)9*O2P*C1P*2)
#define OFF_W3  (OFF_W2 + SZ_W2)

// ---------------- helpers ----------------
// 16-bit A/B fragment per CDNA5 ISA layout: lane l -> row (l&15), k-half (l>>4).
// Lane needs k in {8h..8h+7} U {16+8h..16+8h+7}: two contiguous b128 loads.
__device__ __forceinline__ v16h load_frag(const _Float16* row, int k0, int khalf) {
    v8h lo = *(const v8h*)(row + k0 + 8*khalf);
    v8h hi = *(const v8h*)(row + k0 + 16 + 8*khalf);
    return __builtin_shufflevector(lo, hi, 0,1,2,3,4,5,6,7,8,9,10,11,12,13,14,15);
}

__device__ __forceinline__ v8f wmma_f16(v16h a, v16h b, v8f c) {
    return __builtin_amdgcn_wmma_f32_16x16x32_f16(false, a, false, b, (short)0, c,
                                                  false, false);
}

// ---------------- zero-fill scratch ----------------
__global__ void fill_zero16(uint4* p, long n16) {
    long i = (long)blockIdx.x * blockDim.x + threadIdx.x;
    if (i < n16) { uint4 z = {0u,0u,0u,0u}; p[i] = z; }
}

// ---------------- l1 = sum_d |phi[d,c]| + 1e-12 ----------------
__global__ void l1_kernel(const float* __restrict__ phi, float* __restrict__ l1) {
    int c = blockIdx.x * blockDim.x + threadIdx.x;
    if (c >= C0P) return;
    float s = 0.f;
    if (c < C0)
        for (int d = 0; d < 192; ++d) s += fabsf(phi[(size_t)d*C0 + c]);
    l1[c] = s + 1e-12f;
}

// ---------------- weight prepack (f32 -> f16, K-major rows) ----------------
__global__ void pack_w1(const float* __restrict__ w1, _Float16* __restrict__ wp) {
    long i = (long)blockIdx.x * 256 + threadIdx.x;       // [4][O1P][K1]
    if (i >= (long)4*O1P*K1) return;
    int k  = (int)(i & (K1 - 1));
    int o  = (int)((i >> 11) % O1P);
    int ab = (int)(i / ((long)O1P * K1));
    int c = k & 511, tap = k >> 9;
    int tt = tap >> 1, uu = tap & 1;
    int a = ab >> 1, b = ab & 1;
    float v = 0.f;
    if (c < C0 && o < O1)
        v = w1[(((size_t)c*O1 + o)*4 + (a + 2*tt))*4 + (b + 2*uu)];
    wp[i] = (_Float16)v;
}

__global__ void pack_w2(const float* __restrict__ w2, _Float16* __restrict__ wp) {
    long i = (long)blockIdx.x * 256 + threadIdx.x;       // [9][O2P][C1P]
    if (i >= (long)9*O2P*C1P) return;
    int c   = (int)(i % C1P);
    int o   = (int)((i / C1P) % O2P);
    int tap = (int)(i / ((long)C1P * O2P));
    int kh = tap / 3, kw = tap % 3;
    float v = 0.f;
    if (c < O1 && o < O2)
        v = w2[(((size_t)c*O2 + o)*3 + kh)*3 + kw];
    wp[i] = (_Float16)v;
}

__global__ void pack_w3(const float* __restrict__ w3, _Float16* __restrict__ wp) {
    long i = (long)blockIdx.x * 256 + threadIdx.x;       // [9][O3P][C2P]
    if (i >= (long)9*O3P*C2P) return;
    int c   = (int)(i & (C2P - 1));
    int o   = (int)((i >> 7) & 15);
    int tap = (int)(i >> 11);
    int kh = tap / 3, kw = tap % 3;
    float v = 0.f;
    if (c < O2 && o < 3)
        v = w3[(((size_t)c*3 + o)*3 + kh)*3 + kw];
    wp[i] = (_Float16)v;
}

// ---------------- top-30 |x| select + ternary saturation -> f16 z ----------------
__global__ void __launch_bounds__(256)
topk_sat(const float* __restrict__ x, const float* __restrict__ l1,
         const float* __restrict__ jumpp, _Float16* __restrict__ z) {
    __shared__ int cnt;
    int pix = blockIdx.x;                 // 0..32767 = n*256 + i*16 + j
    int n = pix >> 8, ij = pix & 255;
    int i = ij >> 4, j = ij & 15;
    int t = threadIdx.x;
    int c0 = t, c1 = t + 256;
    const float* xb = x + (size_t)n * C0 * 256 + ij;
    float x0 = xb[(size_t)c0 * 256];
    float x1 = (c1 < C0) ? xb[(size_t)c1 * 256] : 0.f;
    unsigned u0 = __float_as_uint(fabsf(x0));
    unsigned u1 = __float_as_uint(fabsf(x1));
    // radix binary search for the 30th-largest |x| bit pattern
    unsigned cur = 0u;
    for (int bit = 30; bit >= 0; --bit) {
        unsigned T = cur | (1u << bit);
        __syncthreads();
        if (t == 0) cnt = 0;
        __syncthreads();
        int loc = (u0 >= T) + (u1 >= T);
        if (loc) atomicAdd(&cnt, loc);
        __syncthreads();
        if (cnt >= TOPK) cur = T;
    }
    float jmp = jumpp[0];
    _Float16* zb = z + (((size_t)n*ZW + (i + 1))*ZW + (j + 1)) * C0P;
    {   // channel c0 < 500 always
        float l = l1[c0], thr = jmp * l;
        float q = 0.f;
        if (u0 >= cur) q = (x0 > thr) ? l : ((x0 < -thr) ? -l : 0.f);
        zb[c0] = (_Float16)q;
    }
    {   // channel c1 in [256,512): covers pads 500..511 with zeros
        float q = 0.f;
        if (c1 < C0) {
            float l = l1[c1], thr = jmp * l;
            if (u1 >= cur) q = (x1 > thr) ? l : ((x1 < -thr) ? -l : 0.f);
        }
        zb[c1] = (_Float16)q;
    }
}

// ---------------- conv1: stride-2 transpose conv as 4 parity GEMMs ----------------
// Register-blocked 2(M) x 4(N) tiles per wave.
__global__ void __launch_bounds__(32)
conv1_gemm(const _Float16* __restrict__ z, const _Float16* __restrict__ wp,
           const float* __restrict__ b1, _Float16* __restrict__ h1) {
    int lane = threadIdx.x;
    int bx = blockIdx.x;                  // n*10 + mblock (19 M-tiles, MB=2)
    int n = bx / 10, mb = bx % 10;
    int tn0 = blockIdx.y * 4;             // N-tile base (19 N-tiles, NB=4)
    int ab = blockIdx.z;                  // parity
    int a = ab >> 1, b = ab & 1;
    int rc = lane & 15, khf = lane >> 4;  // A-row / B-col, k-half

    int tms[2]; tms[0] = 2*mb; tms[1] = 2*mb + 1 > 18 ? 18 : 2*mb + 1;
    int Ps[2], Qs[2];
#pragma unroll
    for (int s = 0; s < 2; ++s) {
        int idx = tms[s]*16 + rc; if (idx > 288) idx = 288;
        Ps[s] = idx / 17; Qs[s] = idx % 17;
    }
    const _Float16* brow[4];
#pragma unroll
    for (int jn = 0; jn < 4; ++jn) {
        int t = tn0 + jn; if (t > 18) t = 18;   // clamp: dup tiles recompute identical values
        brow[jn] = wp + ((size_t)ab*O1P + (t*16 + rc)) * K1;
    }

    v8f acc[2][4] = {};
    for (int tap = 0; tap < 4; ++tap) {
        int tt = tap >> 1, uu = tap & 1;
        const _Float16* ar[2];
#pragma unroll
        for (int s = 0; s < 2; ++s)
            ar[s] = z + (((size_t)n*ZW + (Ps[s] - tt + 1))*ZW + (Qs[s] - uu + 1)) * C0P;
        for (int k0 = 0; k0 < C0P; k0 += 32) {
            v16h af0 = load_frag(ar[0], k0, khf);
            v16h af1 = load_frag(ar[1], k0, khf);
#pragma unroll
            for (int jn = 0; jn < 4; ++jn) {
                v16h bf = load_frag(brow[jn] + tap*C0P, k0, khf);
                acc[0][jn] = wmma_f16(af0, bf, acc[0][jn]);
                acc[1][jn] = wmma_f16(af1, bf, acc[1][jn]);
            }
        }
    }
#pragma unroll
    for (int jn = 0; jn < 4; ++jn) {
        int t = tn0 + jn; if (t > 18) t = 18;
        int o = t*16 + rc;
        float bias = (o < O1) ? b1[o] : 0.f;
#pragma unroll
        for (int s = 0; s < 2; ++s) {
#pragma unroll
            for (int r = 0; r < 8; ++r) {
                int m = r + 8*khf;
                int id2 = tms[s]*16 + m;
                if (id2 <= 288 && o < O1) {
                    int Pp = id2 / 17, Qq = id2 % 17;
                    int p = 2*Pp + a, q = 2*Qq + b;
                    float v = acc[s][jn][r] + bias; v = v > 0.f ? v : 0.f;
                    h1[(((size_t)n*H1P + (p + 2))*H1P + (q + 2)) * C1P + o] = (_Float16)v;
                }
            }
        }
    }
}

// ---------------- conv2: 3x3 full correlation, 9-tap accumulated GEMM ----------------
// Register-blocked 2(M) x 4(N) tiles per wave.
__global__ void __launch_bounds__(32)
conv2_gemm(const _Float16* __restrict__ h1, const _Float16* __restrict__ wp,
           const float* __restrict__ b2, _Float16* __restrict__ h2) {
    int lane = threadIdx.x;
    int bx = blockIdx.x;                  // n*41 + mblock (81 M-tiles, MB=2)
    int n = bx / 41, mb = bx % 41;
    int tn0 = blockIdx.y * 4;             // 7 N-tiles, NB=4 -> grid.y = 2
    int rc = lane & 15, khf = lane >> 4;

    int tms[2]; tms[0] = 2*mb; tms[1] = 2*mb + 1 > 80 ? 80 : 2*mb + 1;
    int ps[2], qs[2];
#pragma unroll
    for (int s = 0; s < 2; ++s) {
        int idx = tms[s]*16 + rc;
        ps[s] = idx / H2; qs[s] = idx % H2;
    }
    const _Float16* brow[4];
#pragma unroll
    for (int jn = 0; jn < 4; ++jn) {
        int t = tn0 + jn; if (t > 6) t = 6;
        brow[jn] = wp + (size_t)(t*16 + rc) * C1P;
    }

    v8f acc[2][4] = {};
    for (int tap = 0; tap < 9; ++tap) {
        int kh = tap / 3, kw = tap % 3;
        const _Float16* ar[2];
#pragma unroll
        for (int s = 0; s < 2; ++s) {
            ar[s] = h1 + (((size_t)n*H1P + (ps[s] - kh + 2))*H1P + (qs[s] - kw + 2)) * C1P;
            __builtin_prefetch(ar[s], 0, 1);
        }
        const _Float16* bt = wp + (size_t)tap * O2P * C1P;
        for (int k0 = 0; k0 < C1P; k0 += 32) {
            v16h af0 = load_frag(ar[0], k0, khf);
            v16h af1 = load_frag(ar[1], k0, khf);
#pragma unroll
            for (int jn = 0; jn < 4; ++jn) {
                v16h bf = load_frag(brow[jn] + (bt - wp), k0, khf);
                acc[0][jn] = wmma_f16(af0, bf, acc[0][jn]);
                acc[1][jn] = wmma_f16(af1, bf, acc[1][jn]);
            }
        }
    }
#pragma unroll
    for (int jn = 0; jn < 4; ++jn) {
        int t = tn0 + jn; if (t > 6) t = 6;
        int o = t*16 + rc;
        float bias = (o < O2) ? b2[o] : 0.f;
#pragma unroll
        for (int s = 0; s < 2; ++s) {
#pragma unroll
            for (int r = 0; r < 8; ++r) {
                int m = r + 8*khf;
                int id2 = tms[s]*16 + m;
                if (o < O2) {
                    int pp = id2 / H2, qq = id2 % H2;
                    float v = acc[s][jn][r] + bias; v = v > 0.f ? v : 0.f;
                    h2[(((size_t)n*H2 + pp)*H2 + qq) * C2P + o] = (_Float16)v;
                }
            }
        }
    }
}

// ---------------- conv3: 3x3, middle 32x32 only, N padded to 16 ----------------
// Register-blocked 4(M) x 1(N) tiles per wave.
__global__ void __launch_bounds__(32)
conv3_gemm(const _Float16* __restrict__ h2, const _Float16* __restrict__ wp,
           const float* __restrict__ b3, float* __restrict__ out) {
    int lane = threadIdx.x;
    int bx = blockIdx.x;                  // n*16 + mblock (64 M-tiles, MB=4)
    int n = bx >> 4, mb = bx & 15;
    int rc = lane & 15, khf = lane >> 4;

    int py[4], qx[4];
#pragma unroll
    for (int s = 0; s < 4; ++s) {
        int idx = (mb*4 + s)*16 + rc;     // < 1024
        py[s] = (idx >> 5) + 3; qx[s] = (idx & 31) + 3;   // crop offset 3
    }
    const _Float16* brow = wp + (size_t)rc * C2P;

    v8f acc[4] = {};
    for (int tap = 0; tap < 9; ++tap) {
        int kh = tap / 3, kw = tap % 3;
        const _Float16* ar[4];
#pragma unroll
        for (int s = 0; s < 4; ++s) {
            ar[s] = h2 + (((size_t)n*H2 + (py[s] - kh))*H2 + (qx[s] - kw)) * C2P;
            __builtin_prefetch(ar[s], 0, 1);
        }
        const _Float16* bt = brow + (size_t)tap * O3P * C2P;
        for (int k0 = 0; k0 < C2P; k0 += 32) {
            v16h bf = load_frag(bt, k0, khf);
#pragma unroll
            for (int s = 0; s < 4; ++s)
                acc[s] = wmma_f16(load_frag(ar[s], k0, khf), bf, acc[s]);
        }
    }
    int o = rc;
    float bias = (o < 3) ? b3[o] : 0.f;
#pragma unroll
    for (int s = 0; s < 4; ++s) {
#pragma unroll
        for (int r = 0; r < 8; ++r) {
            int m = r + 8*khf;
            int id2 = (mb*4 + s)*16 + m;
            int y2 = id2 >> 5, x2 = id2 & 31;
            if (o < 3) {
                float v = acc[s][r] + bias; v = v > 0.f ? v : 0.f;
                out[(((size_t)n*3 + o)*32 + y2)*32 + x2] = v;
            }
        }
    }
}

// ---------------- launcher ----------------
extern "C" void kernel_launch(void* const* d_in, const int* in_sizes, int n_in,
                              void* d_out, int out_size, void* d_ws, size_t ws_size,
                              hipStream_t stream) {
    (void)in_sizes; (void)n_in; (void)out_size; (void)ws_size;
    const float* x    = (const float*)d_in[0];
    const float* phi  = (const float*)d_in[1];
    const float* jump = (const float*)d_in[2];
    const float* w1   = (const float*)d_in[3];
    const float* b1   = (const float*)d_in[4];
    const float* w2   = (const float*)d_in[5];
    const float* b2   = (const float*)d_in[6];
    const float* w3   = (const float*)d_in[7];
    const float* b3   = (const float*)d_in[8];

    char* ws = (char*)d_ws;
    float*    l1  = (float*)(ws + OFF_L1);
    _Float16* z   = (_Float16*)(ws + OFF_Z);
    _Float16* h1  = (_Float16*)(ws + OFF_H1);
    _Float16* h2  = (_Float16*)(ws + OFF_H2);
    _Float16* wp1 = (_Float16*)(ws + OFF_W1);
    _Float16* wp2 = (_Float16*)(ws + OFF_W2);
    _Float16* wp3 = (_Float16*)(ws + OFF_W3);

    // zero the contiguous z/h1/h2 region (halos + channel pads must be 0)
    long n16 = (long)((SZ_Z + SZ_H1 + SZ_H2) / 16);
    fill_zero16<<<(unsigned)((n16 + 255) / 256), 256, 0, stream>>>((uint4*)(ws + OFF_Z), n16);

    l1_kernel<<<2, 256, 0, stream>>>(phi, l1);
    pack_w1<<<(unsigned)(((long)4*O1P*K1 + 255) / 256), 256, 0, stream>>>(w1, wp1);
    pack_w2<<<(unsigned)(((long)9*O2P*C1P + 255) / 256), 256, 0, stream>>>(w2, wp2);
    pack_w3<<<(unsigned)(((long)9*O3P*C2P + 255) / 256), 256, 0, stream>>>(w3, wp3);

    topk_sat<<<B_ * 256, 256, 0, stream>>>(x, l1, jump, z);

    conv1_gemm<<<dim3(B_ * 10, 5, 4), 32, 0, stream>>>(z, wp1, b1, h1);
    conv2_gemm<<<dim3(B_ * 41, 2, 1), 32, 0, stream>>>(h1, wp2, b2, h2);
    conv3_gemm<<<dim3(B_ * 16, 1, 1), 32, 0, stream>>>(h2, wp3, b3, (float*)d_out);
}
